// ISTFT_55130200212249
// MI455X (gfx1250) — matrix operand order repaired and checked by reading, not compiled, not run on hardware
//
#include <hip/hip_runtime.h>
#include <hip/hip_bf16.h>
#include <math.h>

typedef _Float16 v16h __attribute__((ext_vector_type(16)));
typedef _Float16 h8   __attribute__((ext_vector_type(8)));
typedef float    v8f  __attribute__((ext_vector_type(8)));

#define NBATCH   8
#define TT       4096
#define NFREQ    513
#define NFFT     1024
#define HOP      256
#define WINL     1024
#define OUT_FULL ((TT - 1) * HOP + WINL)   /* 1049344 */
#define PAD      384
#define OUT_TRIM 1048576                   /* per batch, after trim */
#define TILE_N   64
#define TILE_T   64
#define KCHUNKS  32                        /* 32 chunks x 32 = K=1024 */
#define AROW     40                        /* LDS row stride (halves): 80B = 20 banks, conflict-free */
#define BROW     40

// ---------------------------------------------------------------- pass 1: zero accumulator
__global__ void istft_zero(float* __restrict__ ws, long long n) {
    long long i = (long long)blockIdx.x * 256 + threadIdx.x;
    if (i < n) ws[i] = 0.0f;
}

// ---------------------------------------------------------------- pass 2: WMMA GEMM + windowed overlap-add
__global__ __launch_bounds__(256) void istft_gemm(const float* __restrict__ sre,
                                                  const float* __restrict__ sim,
                                                  const float* __restrict__ win,
                                                  float* __restrict__ acc) {
    __shared__ _Float16 Alds[TILE_N * AROW];   // basis panel, n-major   (5 KB)
    __shared__ _Float16 Blds[TILE_T * BROW];   // spectrum panel, t-major (5 KB)

    const int tid = threadIdx.x;
    const int b   = blockIdx.z;
    const int t0  = blockIdx.x * TILE_T;
    const int n0  = blockIdx.y * TILE_N;

    // ---- A-staging identity: each thread owns (n_local, 4 consecutive freqs)
    const int an    = tid & 63;    // n_local 0..63
    const int ag    = tid >> 6;    // freq group 0..3 (4 freqs each -> 16 freqs/chunk)
    const int nglob = n0 + an;

    // phasors for f_j = 1 + 4*ag + j at chunk 0; advanced by delta-f = 16 per chunk
    float pc[4], ps[4];
    const float w0 = 6.28318530717958647692f / 1024.0f * (float)nglob;
    #pragma unroll
    for (int j = 0; j < 4; ++j) {
        __sincosf(w0 * (float)(1 + 4 * ag + j), &ps[j], &pc[j]);
    }
    float dc, dsn;
    __sincosf(w0 * 16.0f, &dsn, &dc);

    // ---- B-staging identity: each thread owns (t_local, 8 consecutive K rows = 4 freq re/im pairs)
    const int btl   = tid & 63;    // t_local 0..63
    const int bg    = tid >> 6;    // row group 0..3 (rows 8*bg..8*bg+7)
    const int tglob = t0 + btl;

    // ---- compute identity: TWO 16x16 tiles per wave (shared A fragment)
    const int lane  = tid & 31;
    const int wave  = tid >> 5;
    const int nsub  = wave & 3;     // 4 n-subtiles
    const int tbase = wave >> 2;    // 0..1 -> t-subtiles 2*tbase, 2*tbase+1
    const int m     = lane & 15;
    const int h     = lane >> 4;

    v8f c0 = {0.f, 0.f, 0.f, 0.f, 0.f, 0.f, 0.f, 0.f};
    v8f c1 = {0.f, 0.f, 0.f, 0.f, 0.f, 0.f, 0.f, 0.f};

    for (int ck = 0; ck < KCHUNKS; ++ck) {
        // ---------------- stage A (basis) via phasor recurrence
        h8 av;
        #pragma unroll
        for (int j = 0; j < 4; ++j) {
            av[2 * j]     = (_Float16)pc[j];     // cos coeff
            av[2 * j + 1] = (_Float16)(-ps[j]);  // -sin coeff
            float nc  = pc[j] * dc - ps[j] * dsn;
            float nsv = ps[j] * dc + pc[j] * dsn;
            pc[j] = nc; ps[j] = nsv;
        }
        if (ck == KCHUNKS - 1 && ag == 3) {
            // kk = 1022/1023: DC and Nyquist columns (scale 1/N = 0.5 * 2/N)
            av[6] = (_Float16)0.5f;
            av[7] = (nglob & 1) ? (_Float16)(-0.5f) : (_Float16)0.5f;
        }
        *(h8*)&Alds[an * AROW + 8 * ag] = av;   // 16B aligned -> ds_store_b128

        // ---------------- stage B (spectrum), coalesced along t; 8 rows/thread
        {
            const int f0 = 16 * ck + 1 + 4 * bg;   // rows kk=32ck+8bg.. = (re,im) of f0..f0+3
            const float* r0 = sre + ((size_t)b * NFREQ + f0) * TT + tglob;
            const float* i0 = sim + ((size_t)b * NFREQ + f0) * TT + tglob;
            h8 bv;
            #pragma unroll
            for (int j = 0; j < 4; ++j) {          // f0+3 <= 512 always: in-bounds
                bv[2 * j]     = (_Float16)r0[(size_t)j * TT];
                bv[2 * j + 1] = (_Float16)i0[(size_t)j * TT];
            }
            if (ck == KCHUNKS - 1 && bg == 3) {
                // kk 1022 -> Xr[0] ; kk 1023 -> Xr[512]
                bv[6] = (_Float16)sre[((size_t)b * NFREQ + 0)   * TT + tglob];
                bv[7] = (_Float16)sre[((size_t)b * NFREQ + 512) * TT + tglob];
            }
            *(h8*)&Blds[btl * BROW + 8 * bg] = bv; // 16B aligned -> ds_store_b128

            if (ck + 1 < KCHUNKS) {                // global_prefetch_b8 next chunk (f += 16)
                __builtin_prefetch(r0 + (size_t)16 * TT, 0, 0);
                __builtin_prefetch(i0 + (size_t)16 * TT, 0, 0);
            }
        }
        __syncthreads();

        // ---------------- fragments + 2x WMMA (A reused across both t-subtiles)
        v16h a, bf0, bf1;
        // A 16-bit layout: lane(m,h): a[0..7] = K h*8+0..7 ; a[8..15] = K 16+h*8+0..7
        *(h8*)&a       = *(const h8*)&Alds[(nsub * 16 + m) * AROW + h * 8];
        *((h8*)&a + 1) = *(const h8*)&Alds[(nsub * 16 + m) * AROW + 16 + h * 8];
        // B 16-bit layout: lane(col=m,h): b[0..7] = K h*8+0..7 ; b[8..15] = K 16+h*8+0..7
        {
            const int col0 = (tbase * 2 + 0) * 16 + m;
            const int col1 = (tbase * 2 + 1) * 16 + m;
            *(h8*)&bf0       = *(const h8*)&Blds[col0 * BROW + h * 8];
            *((h8*)&bf0 + 1) = *(const h8*)&Blds[col0 * BROW + 16 + h * 8];
            *(h8*)&bf1       = *(const h8*)&Blds[col1 * BROW + h * 8];
            *((h8*)&bf1 + 1) = *(const h8*)&Blds[col1 * BROW + 16 + h * 8];
        }
        c0 = __builtin_amdgcn_wmma_f32_16x16x32_f16(false, a, false, bf0,
                                                    (short)0, c0, false, false);
        c1 = __builtin_amdgcn_wmma_f32_16x16x32_f16(false, a, false, bf1,
                                                    (short)0, c1, false, false);
        __syncthreads();
    }

    // ---------------- epilogue: scale, window, overlap-add (4 contributions/sample)
    const float scale = 2.0f / 1024.0f;
    float wreg[8];
    #pragma unroll
    for (int v = 0; v < 8; ++v)
        wreg[v] = scale * win[n0 + nsub * 16 + h * 8 + v];

    #pragma unroll
    for (int ts = 0; ts < 2; ++ts) {
        const v8f cc   = ts ? c1 : c0;
        const int tcol = t0 + (tbase * 2 + ts) * 16 + m;
        float* dst = acc + (size_t)b * OUT_FULL + (size_t)tcol * HOP
                         + (size_t)(n0 + nsub * 16 + h * 8);
        #pragma unroll
        for (int v = 0; v < 8; ++v)
            atomicAdd(dst + v, cc[v] * wreg[v]);   // C layout: M = v + 8h
    }
}

// ---------------------------------------------------------------- pass 3: envelope normalize + trim
__global__ void istft_norm(const float* __restrict__ acc,
                           const float* __restrict__ win,
                           float* __restrict__ out) {
    long long i = (long long)blockIdx.x * 256 + threadIdx.x;
    const long long total = (long long)NBATCH * OUT_TRIM;
    if (i >= total) return;
    const int bidx = (int)(i / OUT_TRIM);
    const int pos  = (int)(i % OUT_TRIM);
    const int s    = pos + PAD;
    const int nb   = s & (HOP - 1);
    const int tmx  = s >> 8;
    float env = 0.0f;
    #pragma unroll
    for (int j = 0; j < 4; ++j) {
        const int t = tmx - j;
        if (t >= 0 && t < TT) {
            const float w = win[nb + j * HOP];
            env += w * w;
        }
    }
    out[i] = acc[(size_t)bidx * OUT_FULL + s] / (env + 1e-11f);
}

// ----------------------------------------------------------------
extern "C" void kernel_launch(void* const* d_in, const int* in_sizes, int n_in,
                              void* d_out, int out_size, void* d_ws, size_t ws_size,
                              hipStream_t stream) {
    (void)in_sizes; (void)n_in; (void)out_size; (void)ws_size;
    const float* sre = (const float*)d_in[0];
    const float* sim = (const float*)d_in[1];
    const float* win = (const float*)d_in[2];
    float* out = (float*)d_out;
    float* acc = (float*)d_ws;   // needs 8 * 1049344 * 4 B = 33.6 MB

    const long long accN = (long long)NBATCH * OUT_FULL;
    istft_zero<<<(int)((accN + 255) / 256), 256, 0, stream>>>(acc, accN);

    dim3 grid(TT / TILE_T, NFFT / TILE_N, NBATCH);   // 64 x 16 x 8
    istft_gemm<<<grid, 256, 0, stream>>>(sre, sim, win, acc);

    const long long outN = (long long)NBATCH * OUT_TRIM;
    istft_norm<<<(int)((outN + 255) / 256), 256, 0, stream>>>(acc, win, out);
}
